// ScatteringTransformFast_90675349553195
// MI455X (gfx1250) — matrix operand order repaired and checked by reading, not compiled, not run on hardware
//
#include <hip/hip_runtime.h>
#include <hip/hip_bf16.h>
#include <math.h>
#include <stdint.h>

// ---------------------------------------------------------------------------
// Scattering transform on MI455X (gfx1250): DFT-as-matmul on bf16x3 WMMA.
//
//  fft2/ifft2 = row pass (X*W) then col pass (W*X); W = DFT matrix (symmetric,
//  1/N per inverse pass). Each real GEMM runs as bf16x3 split precision on
//  V_WMMA_F32_16X16X32_BF16 (hi*hi + hi*lo + lo*hi, f32 accumulate): ~fp32
//  accuracy at the chip's bf16 matrix rate.
//
//  Complex product uses TWO accumulator chains per subtile:
//    im = Ar*Bi + Ai*Br; re = Ar*Br + (-Ai)*Bi (pre-negated A im planes;
//    bf16 negation is an exact sign flip).
//
//  W is symmetric, so the B-operand pair-pack equals the A-operand pair-pack:
//  ONE packed table serves both sides, and both LDS tiles use row-contiguous
//  fragment layout (2 x ds_load_b128 per fragment, no cross-register movs).
//  The shared W operand is staged by async DMA (global_load_async_to_lds_b128
//  / s_wait_asynccnt) with zero conversion; the per-batch f32 data operand is
//  split/packed at staging (filter multiply fused in fp32). Epilogue fuses
//  |.| + scaled mean (atomicAdd into d_out).
// ---------------------------------------------------------------------------

typedef float  v8f  __attribute__((ext_vector_type(8)));
typedef __bf16 v16bf __attribute__((ext_vector_type(16)));
typedef int    v4i  __attribute__((ext_vector_type(4)));

__device__ __forceinline__ v8f wmma_bf16(v16bf a, v16bf b, v8f c) {
  return __builtin_amdgcn_wmma_f32_16x16x32_bf16(false, a, false, b, (short)0,
                                                 c, false, false);
}

#if __has_builtin(__builtin_amdgcn_global_load_async_to_lds_b128) && \
    __has_builtin(__builtin_amdgcn_s_wait_asynccnt)
#define ASYNC_LDS 1
typedef __attribute__((address_space(1))) v4i* as1_v4i;
typedef __attribute__((address_space(3))) v4i* as3_v4i;
__device__ __forceinline__ as1_v4i g_v4i(const void* p) {
  return (as1_v4i)(uintptr_t)p;
}
__device__ __forceinline__ as3_v4i l_v4i(void* p) {
  return (as3_v4i)(uint32_t)(uintptr_t)p;   // low 32 bits = LDS offset
}
#endif

#define TWO_PI 6.2831853071795864769f
#define SGNMASK 0x80008000u

// ---- bf16 split helpers ----------------------------------------------------
__device__ __forceinline__ void split_bf(float x, unsigned short& h,
                                         unsigned short& l) {
  __bf16 hb = (__bf16)x;
  __bf16 lb = (__bf16)(x - (float)hb);
  h = __builtin_bit_cast(unsigned short, hb);
  l = __builtin_bit_cast(unsigned short, lb);
}
__device__ __forceinline__ unsigned pack2(unsigned short e0, unsigned short e1) {
  return (unsigned)e0 | ((unsigned)e1 << 16);   // K-even in [15:0]
}

// Packed symmetric-DFT table: plane[m][q] packs {W[m][2q], W[m][2q+1]}.
// Serves as A-operand (rows = output m) AND B-operand (rows = column n, by
// symmetry). 6 planes: reHi, reLo, imHi, imLo, -imHi, -imLo.
struct WPack { const unsigned *reHi, *reLo, *imHi, *imLo, *nimHi, *nimLo; };

// --------------------------- table generation ------------------------------
__global__ void gen_dft_packed(unsigned* __restrict__ reHi,
                               unsigned* __restrict__ reLo,
                               unsigned* __restrict__ imHi,
                               unsigned* __restrict__ imLo,
                               unsigned* __restrict__ nimHi,
                               unsigned* __restrict__ nimLo,
                               int N, float sign, float scale) {
  int half = N >> 1;
  int i = blockIdx.x * blockDim.x + threadIdx.x;
  if (i >= N * half) return;
  int a = i / half, k0 = 2 * (i % half);
  int t0 = (int)(((long long)k0 * a) % N);
  int t1 = (int)(((long long)(k0 + 1) * a) % N);
  float ang0 = sign * TWO_PI * (float)t0 / (float)N;
  float ang1 = sign * TWO_PI * (float)t1 / (float)N;
  float e0r = cosf(ang0) * scale, e0i = sinf(ang0) * scale;
  float e1r = cosf(ang1) * scale, e1i = sinf(ang1) * scale;
  unsigned short h0, l0, h1, l1;
  split_bf(e0r, h0, l0); split_bf(e1r, h1, l1);
  reHi[i] = pack2(h0, h1); reLo[i] = pack2(l0, l1);
  split_bf(e0i, h0, l0); split_bf(e1i, h1, l1);
  unsigned ih = pack2(h0, h1), il = pack2(l0, l1);
  imHi[i] = ih; imLo[i] = il;
  nimHi[i] = ih ^ SGNMASK; nimLo[i] = il ^ SGNMASK;
}

__global__ void zero_kernel(float* __restrict__ p, int n) {
  int i = blockIdx.x * blockDim.x + threadIdx.x;
  if (i < n) p[i] = 0.0f;
}

// ------------------------------- S0 ----------------------------------------
__global__ void s0_mean_kernel(const float* __restrict__ in, float* __restrict__ out) {
  __shared__ float red[256];
  int b = blockIdx.x;
  const float* p = in + (size_t)b * 65536;
  float s = 0.0f;
  for (int i = threadIdx.x; i < 65536; i += 256) s += p[i];
  red[threadIdx.x] = s;
  __syncthreads();
  for (int o = 128; o > 0; o >>= 1) {
    if (threadIdx.x < (unsigned)o) red[threadIdx.x] += red[threadIdx.x + o];
    __syncthreads();
  }
  if (threadIdx.x == 0) out[b] = red[0] * (1.0f / 65536.0f);
}

// --------------------------- spectral corner cut ---------------------------
__global__ void cut_corners_kernel(const float* __restrict__ Sre,
                                   const float* __restrict__ Sim,
                                   float* __restrict__ Dre, float* __restrict__ Dim,
                                   int Nsrc, int c, int nbatch) {
  int total = nbatch * c * c;
  for (int i = blockIdx.x * blockDim.x + threadIdx.x; i < total;
       i += gridDim.x * blockDim.x) {
    int z = i / (c * c);
    int rc = i % (c * c);
    int r = rc / c, col = rc % c;
    int dx = c >> 1;
    int sr = (r < dx) ? r : (Nsrc - c + r);
    int sc = (col < dx) ? col : (Nsrc - c + col);
    size_t si = (size_t)z * Nsrc * Nsrc + (size_t)sr * Nsrc + sc;
    Dre[i] = Sre[si];
    Dim[i] = Sim[si];
  }
}

// ---- WMMA fragment loads (row-contiguous => 2 x ds_load_b128 each) --------
union BFrag { unsigned u[8]; v16bf v; };

// A 16x32 bf16: VGPR v, lane(lo,hi): K = 2*(v&3) + 8*hi + 16*(v>>2) (+1)
__device__ __forceinline__ v16bf load_afrag(const unsigned (*pl)[20], int m, int hi) {
  BFrag f;
#pragma unroll
  for (int v = 0; v < 8; ++v) f.u[v] = pl[m][(v & 3) + 4 * hi + 8 * (v >> 2)];
  return f.v;
}
// B 32x16 bf16: VGPR v, lane(lo,hi): K = 2*(v&3) + 8*hi + 16*(v>>2) (+1); tile
// stored as [n][pairrow], so lane n reads dwords (v&3) + 4*hi + 8*(v>>2).
__device__ __forceinline__ v16bf load_bfrag(const unsigned (*pl)[20], int n, int hi) {
  BFrag f;
#pragma unroll
  for (int v = 0; v < 8; ++v) f.u[v] = pl[n][(v & 3) + 4 * hi + 8 * (v >> 2)];
  return f.v;
}

// ---------------- generic batched complex GEMM, bf16x3 ---------------------
// dataOnA=1: C[z] = (data[z/Lmod] .* Filt[z%Lmod]) * W     (row pass)
// dataOnA=0: C[z] = W * data[z]                            (col pass, Lmod=1)
// mode: bit0 = store |.| (re=abs, im=0); bit1 = atomicAdd scaled sum of |.|
//       to redBase[(z/Ldiv)*redStride]; neither => store complex.
// Block 128 thr (4 waves), 64x64 tile, wave 32x32; K-chunk 32; 48 bf16 WMMA
// per chunk per wave, 2 accumulator chains per subtile.
__global__ void __launch_bounds__(128)
cgemm_bf16x3(int dataOnA,
             const float* __restrict__ dRe, const float* __restrict__ dIm,
             long strideD, const float2* __restrict__ Filt, int Lmod,
             WPack W,
             float* __restrict__ Cre, float* __restrict__ Cim, long strideC,
             float* __restrict__ redBase, int redStride, int Ldiv, float scale,
             int mode, int M, int N, int K) {
  // A planes: 0=reHi 1=reLo 2=imHi 3=imLo 4=-imHi 5=-imLo ; [plane][m][qpair]
  __shared__ unsigned lA[6][64][20];
  // B planes: 0=reHi 1=reLo 2=imHi 3=imLo ; [plane][n][pairrow]
  __shared__ unsigned lB[4][64][20];

  const int tid = threadIdx.x;
  const int wave = tid >> 5;
  const int lane = tid & 31;
  const int lo = lane & 15;
  const int hi = lane >> 4;

  const int mBase = blockIdx.y * 64;
  const int nBase = blockIdx.x * 64;
  const long z = blockIdx.z;

  const int waveM = (wave >> 1) * 32;
  const int waveN = (wave & 1) * 32;

  const float* dr = dRe + (z / Lmod) * (size_t)strideD;
  const float* di = dIm ? (dIm + (z / Lmod) * (size_t)strideD) : nullptr;
  const float2* Fp = (dataOnA && Filt)
                         ? (Filt + (z % Lmod) * (size_t)M * K) : nullptr;
  const unsigned* wp[6] = {W.reHi, W.reLo, W.imHi, W.imLo, W.nimHi, W.nimLo};
  const int halfK = K >> 1;

  v8f accRe[2][2], accIm[2][2];
#pragma unroll
  for (int mi = 0; mi < 2; ++mi)
#pragma unroll
    for (int ni = 0; ni < 2; ++ni) {
      accRe[mi][ni] = (v8f){0.f,0.f,0.f,0.f,0.f,0.f,0.f,0.f};
      accIm[mi][ni] = (v8f){0.f,0.f,0.f,0.f,0.f,0.f,0.f,0.f};
    }

  for (int kc = 0; kc < K; kc += 32) {
    const int kq = kc >> 1;               // pair index base
    bool asyncUsed = false;

    if (dataOnA) {
      // ---- B = W, symmetric pack rows = column n: async when full ----
#if defined(ASYNC_LDS)
      if (nBase + 64 <= N) {
        asyncUsed = true;
        for (int e = tid; e < 1024; e += 128) {   // 4 planes x 64 rows x 4 b128
          int p = e >> 8, r = (e >> 2) & 63, t = e & 3;
          size_t gi = (size_t)(nBase + r) * halfK + kq + 4 * t;
          __builtin_amdgcn_global_load_async_to_lds_b128(
              g_v4i(wp[p] + gi), l_v4i(&lB[p][r][4 * t]), 0, 0);
        }
      } else
#endif
      {
        for (int e = tid; e < 4096; e += 128) {
          int p = e >> 10, r = (e >> 4) & 63, q = e & 15;
          lB[p][r][q] = (nBase + r < N)
                            ? wp[p][(size_t)(nBase + r) * halfK + kq + q] : 0u;
        }
      }
      // ---- A = data (f32 -> split/pack, 6 planes; filter fused in fp32) ----
      for (int e = tid; e < 1024; e += 128) {     // 64 rows x 16 pairs
        int r = e >> 4, q = e & 15;
        float r0 = 0.f, r1 = 0.f, i0 = 0.f, i1 = 0.f;
        if (mBase + r < M) {
          size_t b0 = (size_t)(mBase + r) * K + kc + 2 * q;
          r0 = dr[b0]; r1 = dr[b0 + 1];
          if (di) { i0 = di[b0]; i1 = di[b0 + 1]; }
          if (Fp) {
            float2 f0 = Fp[b0], f1 = Fp[b0 + 1];
            float t0 = r0 * f0.x - i0 * f0.y; i0 = r0 * f0.y + i0 * f0.x; r0 = t0;
            float t1 = r1 * f1.x - i1 * f1.y; i1 = r1 * f1.y + i1 * f1.x; r1 = t1;
          }
        }
        unsigned short h0, l0w, h1, l1w;
        split_bf(r0, h0, l0w); split_bf(r1, h1, l1w);
        lA[0][r][q] = pack2(h0, h1); lA[1][r][q] = pack2(l0w, l1w);
        split_bf(i0, h0, l0w); split_bf(i1, h1, l1w);
        unsigned ih = pack2(h0, h1), il = pack2(l0w, l1w);
        lA[2][r][q] = ih;            lA[3][r][q] = il;
        lA[4][r][q] = ih ^ SGNMASK;  lA[5][r][q] = il ^ SGNMASK;
      }
    } else {
      // ---- A = W (6 planes): async when full ----
#if defined(ASYNC_LDS)
      if (mBase + 64 <= M) {
        asyncUsed = true;
        for (int e = tid; e < 1536; e += 128) {   // 6 planes x 64 rows x 4 b128
          int p = e >> 8, r = (e >> 2) & 63, t = e & 3;
          size_t gi = (size_t)(mBase + r) * halfK + kq + 4 * t;
          __builtin_amdgcn_global_load_async_to_lds_b128(
              g_v4i(wp[p] + gi), l_v4i(&lA[p][r][4 * t]), 0, 0);
        }
      } else
#endif
      {
        for (int e = tid; e < 6144; e += 128) {
          int p = e >> 10, r = (e >> 4) & 63, q = e & 15;
          lA[p][r][q] = (mBase + r < M)
                            ? wp[p][(size_t)(mBase + r) * halfK + kq + q] : 0u;
        }
      }
      // ---- B = data (f32 -> split/pack into [n][pairrow]) ----
      for (int e = tid; e < 1024; e += 128) {     // 16 pair-rows x 64 cols
        int pr = e >> 6, n0 = e & 63;
        int gn = nBase + n0;
        float r0 = 0.f, r1 = 0.f, i0 = 0.f, i1 = 0.f;
        if (gn < N) {
          size_t b0 = (size_t)(kc + 2 * pr) * N + gn;
          r0 = dr[b0]; r1 = dr[b0 + N];
          if (di) { i0 = di[b0]; i1 = di[b0 + N]; }
        }
        unsigned short h0, l0w, h1, l1w;
        split_bf(r0, h0, l0w); split_bf(r1, h1, l1w);
        lB[0][n0][pr] = pack2(h0, h1); lB[1][n0][pr] = pack2(l0w, l1w);
        split_bf(i0, h0, l0w); split_bf(i1, h1, l1w);
        lB[2][n0][pr] = pack2(h0, h1); lB[3][n0][pr] = pack2(l0w, l1w);
      }
    }

#if defined(ASYNC_LDS)
    if (asyncUsed) __builtin_amdgcn_s_wait_asynccnt(0);
#else
    (void)asyncUsed;
#endif
    __syncthreads();

    // ---- bf16x3 complex MACs: 4 subtiles x 12 = 48 WMMA per chunk ----
    v16bf BrH[2], BrL[2], BiH[2], BiL[2];
#pragma unroll
    for (int ni = 0; ni < 2; ++ni) {
      int n = waveN + ni * 16 + lo;
      BrH[ni] = load_bfrag(lB[0], n, hi);
      BrL[ni] = load_bfrag(lB[1], n, hi);
      BiH[ni] = load_bfrag(lB[2], n, hi);
      BiL[ni] = load_bfrag(lB[3], n, hi);
    }
#pragma unroll
    for (int mi = 0; mi < 2; ++mi) {
      int m = waveM + mi * 16 + lo;
      v16bf ArH = load_afrag(lA[0], m, hi);
      v16bf ArL = load_afrag(lA[1], m, hi);
      v16bf AiH = load_afrag(lA[2], m, hi);
      v16bf AiL = load_afrag(lA[3], m, hi);
      v16bf AnH = load_afrag(lA[4], m, hi);
      v16bf AnL = load_afrag(lA[5], m, hi);
#pragma unroll
      for (int ni = 0; ni < 2; ++ni) {
        v8f r = accRe[mi][ni];
        r = wmma_bf16(ArH, BrH[ni], r);     // + Ar*Br
        r = wmma_bf16(ArH, BrL[ni], r);
        r = wmma_bf16(ArL, BrH[ni], r);
        r = wmma_bf16(AnH, BiH[ni], r);     // - Ai*Bi (pre-negated A planes)
        r = wmma_bf16(AnH, BiL[ni], r);
        r = wmma_bf16(AnL, BiH[ni], r);
        accRe[mi][ni] = r;
        v8f q = accIm[mi][ni];
        q = wmma_bf16(ArH, BiH[ni], q);     // + Ar*Bi
        q = wmma_bf16(ArH, BiL[ni], q);
        q = wmma_bf16(ArL, BiH[ni], q);
        q = wmma_bf16(AiH, BrH[ni], q);     // + Ai*Br
        q = wmma_bf16(AiH, BrL[ni], q);
        q = wmma_bf16(AiL, BrH[ni], q);
        accIm[mi][ni] = q;
      }
    }
    __syncthreads();
  }

  // ---- epilogue ----
  float part = 0.0f;
  size_t cb = (size_t)z * strideC;
#pragma unroll
  for (int mi = 0; mi < 2; ++mi) {
#pragma unroll
    for (int ni = 0; ni < 2; ++ni) {
      int sm = mBase + waveM + mi * 16;     // wave-uniform guards
      int sn = nBase + waveN + ni * 16;
      if (sm >= M || sn >= N) continue;
#pragma unroll
      for (int v = 0; v < 8; ++v) {
        int row = sm + v + 8 * hi;
        int col = sn + lo;
        size_t o = cb + (size_t)row * N + col;
        float re = accRe[mi][ni][v];
        float im = accIm[mi][ni][v];
        if (mode & 1) {
          float a = sqrtf(re * re + im * im);
          Cre[o] = a;
          Cim[o] = 0.0f;
          if (mode & 2) part += a;
        } else if (mode & 2) {
          part += sqrtf(re * re + im * im);
        } else {
          Cre[o] = re;
          Cim[o] = im;
        }
      }
    }
  }
  if (mode & 2) {
    for (int off = 16; off > 0; off >>= 1) part += __shfl_xor(part, off, 32);
    if (lane == 0)
      atomicAdd(redBase + (z / Ldiv) * (size_t)redStride, part * scale);
  }
}

// ---------------------------------------------------------------------------
extern "C" void kernel_launch(void* const* d_in, const int* in_sizes, int n_in,
                              void* d_out, int out_size, void* d_ws, size_t ws_size,
                              hipStream_t stream) {
  (void)in_sizes; (void)n_in; (void)out_size; (void)ws_size;

  const int J = 4, L = 8, SIZE = 256, B = 32;
  const int CUTS[4] = {256, 128, 64, 32};

  const float* input = (const float*)d_in[0];
  const float2* filt[4] = {(const float2*)d_in[1], (const float2*)d_in[2],
                           (const float2*)d_in[3], (const float2*)d_in[4]};
  float* out = (float*)d_out;   // [32] S0 | [32*4] s1 | [32*4*4] s2

  // ---- workspace carve-up (4-byte units) ----
  float* w = (float*)d_ws;
  size_t cur = 0;
  auto alloc  = [&](size_t n) { float* p = w + cur; cur += n; return p; };
  auto allocu = [&](size_t n) { return (unsigned*)alloc(n); };

  // packed symmetric DFT tables: per size, forward (F) and inverse (V)
  WPack Fw[4], Vw[4];
  for (int s = 0; s < 4; ++s) {
    size_t np = (size_t)CUTS[s] * (CUTS[s] / 2);
    WPack* packs[2] = {&Fw[s], &Vw[s]};
    for (int t = 0; t < 2; ++t) {
      unsigned* rh = allocu(np); unsigned* rl = allocu(np);
      unsigned* ih = allocu(np); unsigned* il = allocu(np);
      unsigned* nh = allocu(np); unsigned* nl = allocu(np);
      *packs[t] = WPack{rh, rl, ih, il, nh, nl};
      int nb = (int)((np + 255) / 256);
      float sign  = t ? 1.0f : -1.0f;
      float scale = t ? 1.0f / (float)CUTS[s] : 1.0f;
      gen_dft_packed<<<nb, 256, 0, stream>>>(rh, rl, ih, il, nh, nl,
                                             CUTS[s], sign, scale);
    }
  }

  float* I0Kre = alloc((size_t)B * 65536);
  float* I0Kim = alloc((size_t)B * 65536);
  float* CAre  = alloc((size_t)B * 65536);
  float* CAim  = alloc((size_t)B * 65536);
  float* BAre  = alloc((size_t)B * L * 65536);
  float* BAim  = alloc((size_t)B * L * 65536);
  float* BBre  = alloc((size_t)B * L * 65536);
  float* BBim  = alloc((size_t)B * L * 65536);
  float* CBre  = alloc((size_t)B * L * 16384);
  float* CBim  = alloc((size_t)B * L * 16384);
  float* T3re  = alloc((size_t)B * L * 16384);
  float* T3im  = alloc((size_t)B * L * 16384);

  zero_kernel<<<3, 256, 0, stream>>>(out, 672);
  s0_mean_kernel<<<B, 256, 0, stream>>>(input, out);

  auto grid = [](int M, int N, int nb) {
    return dim3((unsigned)((N + 63) / 64), (unsigned)((M + 63) / 64), (unsigned)nb);
  };

  // ---- I0_k = fft2(input) ----
  cgemm_bf16x3<<<grid(SIZE, SIZE, B), 128, 0, stream>>>(
      1, input, nullptr, (long)SIZE * SIZE, nullptr, 1, Fw[0],
      BAre, BAim, (long)SIZE * SIZE,
      nullptr, 0, 1, 0.0f, 0, SIZE, SIZE, SIZE);
  cgemm_bf16x3<<<grid(SIZE, SIZE, B), 128, 0, stream>>>(
      0, BAre, BAim, (long)SIZE * SIZE, nullptr, 1, Fw[0],
      I0Kre, I0Kim, (long)SIZE * SIZE,
      nullptr, 0, 1, 0.0f, 0, SIZE, SIZE, SIZE);

  for (int j1 = 0; j1 < J; ++j1) {
    int c1 = CUTS[j1];
    long p1 = (long)c1 * c1;
    {
      int total = B * c1 * c1;
      cut_corners_kernel<<<(total + 255) / 256, 256, 0, stream>>>(
          I0Kre, I0Kim, CAre, CAim, SIZE, c1, B);
    }
    // ifft2 of (CA .* filt_j1[l]); batch expands to B*L
    cgemm_bf16x3<<<grid(c1, c1, B * L), 128, 0, stream>>>(
        1, CAre, CAim, p1, filt[j1], L, Vw[j1],
        BAre, BAim, p1, nullptr, 0, 1, 0.0f, 0, c1, c1, c1);
    // col pass: store |I1| + accumulate s1[b][j1]; scale = 1/(65536*L)
    cgemm_bf16x3<<<grid(c1, c1, B * L), 128, 0, stream>>>(
        0, BAre, BAim, p1, nullptr, 1, Vw[j1],
        BBre, BBim, p1,
        out + 32 + j1, J, L, 1.0f / (65536.0f * (float)L), 3, c1, c1, c1);

    if (j1 == J - 1) continue;

    // I1_k = fft2(|I1|)
    cgemm_bf16x3<<<grid(c1, c1, B * L), 128, 0, stream>>>(
        1, BBre, BBim, p1, nullptr, 1, Fw[j1],
        BAre, BAim, p1, nullptr, 0, 1, 0.0f, 0, c1, c1, c1);
    cgemm_bf16x3<<<grid(c1, c1, B * L), 128, 0, stream>>>(
        0, BAre, BAim, p1, nullptr, 1, Fw[j1],
        BBre, BBim, p1, nullptr, 0, 1, 0.0f, 0, c1, c1, c1);

    for (int j2 = j1 + 1; j2 < J; ++j2) {
      int c2 = CUTS[j2];
      long p2 = (long)c2 * c2;
      {
        int total = B * L * c2 * c2;
        cut_corners_kernel<<<(total + 255) / 256, 256, 0, stream>>>(
            BBre, BBim, CBre, CBim, c1, c2, B * L);
      }
      for (int l2 = 0; l2 < L; ++l2) {
        const float2* f2 = filt[j2] + (size_t)l2 * c2 * c2;
        cgemm_bf16x3<<<grid(c2, c2, B * L), 128, 0, stream>>>(
            1, CBre, CBim, p2, f2, 1, Vw[j2],
            T3re, T3im, p2, nullptr, 0, 1, 0.0f, 0, c2, c2, c2);
        // reduce-only epilogue -> s2[b][j1][j2]; scale = 1/(c1^2 * 64)
        cgemm_bf16x3<<<grid(c2, c2, B * L), 128, 0, stream>>>(
            0, T3re, T3im, p2, nullptr, 1, Vw[j2],
            nullptr, nullptr, 0L,
            out + 32 + 128 + j1 * J + j2, J * J, L,
            1.0f / ((float)c1 * (float)c1 * 64.0f), 2, c2, c2, c2);
      }
    }
  }
}
